// TinyMoELayer_9199819948301
// MI455X (gfx1250) — matrix operand order repaired and checked by reading, not compiled, not run on hardware
//
#include <hip/hip_runtime.h>
#include <hip/hip_bf16.h>
#include <stdint.h>

#define H_DIM   1024
#define I_DIM   4096
#define NE      8
#define T_TOK   8192
#define LN_EPS  1e-5f

// TDM-padded LDS tile geometry for the 32x1024 bf16 activation tile:
// rows are 2048B data + 24B pad inserted by TDM after every 1024B stored
// -> row stride 2096B = 524 DW == 12 (mod 64): 16 row-lanes land on 16
//    distinct bank quads => conflict-free ds_load_b128.
#define TA_ROW_STRIDE 2096

typedef __bf16 bf16;
typedef __bf16 bf16x4  __attribute__((ext_vector_type(4)));
typedef __bf16 bf16x16 __attribute__((ext_vector_type(16)));
typedef float  f32x8   __attribute__((ext_vector_type(8)));
typedef unsigned int u32x4 __attribute__((ext_vector_type(4)));
typedef int  i32x4 __attribute__((ext_vector_type(4)));
typedef int  i32x8 __attribute__((ext_vector_type(8)));

union AFrag { bf16x16 v; u32x4 q[2]; };

__device__ __forceinline__ f32x8 wmma_bf16(bf16x16 a, bf16x16 b, f32x8 c) {
    // D = A(16x32) * B(32x16) + C, fp32 accumulate
    return __builtin_amdgcn_wmma_f32_16x16x32_bf16(
        /*neg_a=*/false, a, /*neg_b=*/false, b,
        /*c_mod=*/(short)0, c, /*reuse_a=*/false, /*reuse_b=*/false);
}

__device__ __forceinline__ f32x8 zero8() {
    f32x8 z = {0.f,0.f,0.f,0.f,0.f,0.f,0.f,0.f};
    return z;
}

// async global->LDS copy of 16 bytes; tracked with ASYNCcnt
__device__ __forceinline__ void async_copy_b128(unsigned lds_off, const void* gaddr) {
    asm volatile("global_load_async_to_lds_b128 %0, %1, off"
                 :: "v"(lds_off), "v"(gaddr) : "memory");
}
__device__ __forceinline__ void wait_asynccnt0() {
    asm volatile("s_wait_asynccnt 0x0" ::: "memory");
}
// WGP-scope prefetch (scope 0 pulls into all cache levels incl. WGP$)
__device__ __forceinline__ void prefetch_wgp(const void* p) {
    asm volatile("global_prefetch_b8 %0, off" :: "v"(p));
}

// byte offset of element (row, pos) inside the TDM-padded tile
__device__ __forceinline__ unsigned tA_off(int row, int pos) {
    return (unsigned)(row * TA_ROW_STRIDE + pos * 2 + ((pos >= 512) ? 24 : 0));
}

// ---------------------------------------------------------------- init
__global__ void k_init(int* __restrict__ counts) {
    if (threadIdx.x < NE) counts[threadIdx.x] = 0;
}

// ---------------------------------------------------------------- fp32 -> bf16
__global__ __launch_bounds__(256) void k_cvt(const float* __restrict__ s,
                                             bf16* __restrict__ d, int n) {
    int stride = gridDim.x * blockDim.x * 4;
    for (int i = (blockIdx.x * blockDim.x + threadIdx.x) * 4; i < n; i += stride) {
        float4 f = *(const float4*)(s + i);
        bf16x4 o = { (bf16)f.x, (bf16)f.y, (bf16)f.z, (bf16)f.w };
        *(bf16x4*)(d + i) = o;
    }
}

// ---------------------------------------------------------------- LN1 + attn GEMM + residual
__global__ __launch_bounds__(256) void k_ln_attn(
    const float* __restrict__ x, const float* __restrict__ g1, const float* __restrict__ b1,
    const bf16* __restrict__ Wa, float* __restrict__ hout)
{
    __shared__ bf16 aT[32][H_DIM + 8];          // padded: stride 2064B kills bank conflicts
    const int tile = blockIdx.x;                 // 256 tiles of 32 tokens
    const int tid  = threadIdx.x;
    const int w    = tid >> 5, l = tid & 31;
    const int khalf = l >> 4, arow = l & 15;

    // ---- LN1: wave w owns rows 4w..4w+3
    #pragma unroll
    for (int rr = 0; rr < 4; ++rr) {
        const int row = tile * 32 + w * 4 + rr;
        const float* xr = x + (size_t)row * H_DIM;
        float s = 0.f, s2 = 0.f;
        for (int c = l * 4; c < H_DIM; c += 128) {
            float4 f = *(const float4*)(xr + c);
            s  += f.x + f.y + f.z + f.w;
            s2 += f.x*f.x + f.y*f.y + f.z*f.z + f.w*f.w;
        }
        #pragma unroll
        for (int off = 16; off; off >>= 1) { s += __shfl_xor(s, off); s2 += __shfl_xor(s2, off); }
        const float mean = s * (1.f / H_DIM);
        const float rstd = rsqrtf(s2 * (1.f / H_DIM) - mean * mean + LN_EPS);
        for (int c = l * 4; c < H_DIM; c += 128) {
            float4 f  = *(const float4*)(xr + c);
            float4 gg = *(const float4*)(g1 + c);
            float4 bb = *(const float4*)(b1 + c);
            bf16x4 o = { (bf16)((f.x - mean) * rstd * gg.x + bb.x),
                         (bf16)((f.y - mean) * rstd * gg.y + bb.y),
                         (bf16)((f.z - mean) * rstd * gg.z + bb.z),
                         (bf16)((f.w - mean) * rstd * gg.w + bb.w) };
            *(bf16x4*)(&aT[w * 4 + rr][c]) = o;
        }
    }
    __syncthreads();

    // ---- GEMM: wave w -> output cols [128w, 128w+128)
    f32x8 acc[2][8];
    #pragma unroll
    for (int ms = 0; ms < 2; ++ms)
        #pragma unroll
        for (int n = 0; n < 8; ++n) acc[ms][n] = zero8();

    for (int k = 0; k < H_DIM; k += 32) {
        AFrag a0, a1;
        a0.q[0] = *(const u32x4*)&aT[arow     ][k     + khalf * 16];
        a0.q[1] = *(const u32x4*)&aT[arow     ][k + 8 + khalf * 16];
        a1.q[0] = *(const u32x4*)&aT[16 + arow][k     + khalf * 16];
        a1.q[1] = *(const u32x4*)&aT[16 + arow][k + 8 + khalf * 16];
        #pragma unroll
        for (int n = 0; n < 8; ++n) {
            const int col = w * 128 + n * 16 + (l & 15);
            const bf16* bp = Wa + (size_t)col * H_DIM + k + khalf * 16;
            bf16x16 b = *(const bf16x16*)bp;
            prefetch_wgp(bp + 32);
            acc[0][n] = wmma_bf16(a0.v, b, acc[0][n]);
            acc[1][n] = wmma_bf16(a1.v, b, acc[1][n]);
        }
    }

    // ---- epilogue: h = x + acc
    #pragma unroll
    for (int ms = 0; ms < 2; ++ms)
        #pragma unroll
        for (int n = 0; n < 8; ++n)
            #pragma unroll
            for (int r = 0; r < 8; ++r) {
                const int row = tile * 32 + ms * 16 + khalf * 8 + r;
                const int col = w * 128 + n * 16 + (l & 15);
                hout[(size_t)row * H_DIM + col] = x[(size_t)row * H_DIM + col] + acc[ms][n][r];
            }
}

// ---------------------------------------------------------------- LN2 -> bf16 t
__global__ __launch_bounds__(256) void k_ln2(
    const float* __restrict__ h, const float* __restrict__ g2, const float* __restrict__ b2,
    bf16* __restrict__ t)
{
    const int w = threadIdx.x >> 5, l = threadIdx.x & 31;
    const int row = blockIdx.x * 8 + w;
    const float* hr = h + (size_t)row * H_DIM;
    float s = 0.f, s2 = 0.f;
    for (int c = l * 4; c < H_DIM; c += 128) {
        float4 f = *(const float4*)(hr + c);
        s  += f.x + f.y + f.z + f.w;
        s2 += f.x*f.x + f.y*f.y + f.z*f.z + f.w*f.w;
    }
    #pragma unroll
    for (int off = 16; off; off >>= 1) { s += __shfl_xor(s, off); s2 += __shfl_xor(s2, off); }
    const float mean = s * (1.f / H_DIM);
    const float rstd = rsqrtf(s2 * (1.f / H_DIM) - mean * mean + LN_EPS);
    for (int c = l * 4; c < H_DIM; c += 128) {
        float4 f  = *(const float4*)(hr + c);
        float4 gg = *(const float4*)(g2 + c);
        float4 bb = *(const float4*)(b2 + c);
        bf16x4 o = { (bf16)((f.x - mean) * rstd * gg.x + bb.x),
                     (bf16)((f.y - mean) * rstd * gg.y + bb.y),
                     (bf16)((f.z - mean) * rstd * gg.z + bb.z),
                     (bf16)((f.w - mean) * rstd * gg.w + bb.w) };
        *(bf16x4*)(t + (size_t)row * H_DIM + c) = o;
    }
}

// ---------------------------------------------------------------- gate: top-2 + compaction
__global__ __launch_bounds__(256) void k_gate(
    const bf16* __restrict__ t, const float* __restrict__ gw,
    int* __restrict__ counts, int* __restrict__ idxl, float* __restrict__ wgtl)
{
    const int w = threadIdx.x >> 5, l = threadIdx.x & 31;
    const int tok = blockIdx.x * 8 + w;
    float lg[NE];
    #pragma unroll
    for (int e = 0; e < NE; ++e) lg[e] = 0.f;
    const bf16* tr = t + (size_t)tok * H_DIM;
    for (int c = l * 4; c < H_DIM; c += 128) {
        bf16x4 tv = *(const bf16x4*)(tr + c);
        const float t0 = (float)tv[0], t1 = (float)tv[1], t2 = (float)tv[2], t3 = (float)tv[3];
        #pragma unroll
        for (int e = 0; e < NE; ++e) {
            float4 gv = *(const float4*)(gw + (size_t)e * H_DIM + c);
            lg[e] += t0 * gv.x + t1 * gv.y + t2 * gv.z + t3 * gv.w;
        }
    }
    #pragma unroll
    for (int e = 0; e < NE; ++e)
        #pragma unroll
        for (int off = 16; off; off >>= 1) lg[e] += __shfl_xor(lg[e], off);

    if (l == 0) {
        int i1 = 0;
        #pragma unroll
        for (int e = 1; e < NE; ++e) if (lg[e] > lg[i1]) i1 = e;
        int i2 = (i1 == 0) ? 1 : 0;
        #pragma unroll
        for (int e = 0; e < NE; ++e) if (e != i1 && lg[e] > lg[i2]) i2 = e;
        const float e2 = __expf(lg[i2] - lg[i1]);
        const float inv = 1.f / (1.f + e2);
        const int p1 = atomicAdd(&counts[i1], 1);
        idxl[i1 * T_TOK + p1] = tok;  wgtl[i1 * T_TOK + p1] = inv;
        const int p2 = atomicAdd(&counts[i2], 1);
        idxl[i2 * T_TOK + p2] = tok;  wgtl[i2 * T_TOK + p2] = e2 * inv;
    }
}

// ---------------------------------------------------------------- routed expert FFN
__global__ __launch_bounds__(256) void k_moe(
    const bf16* __restrict__ t,
    const bf16* __restrict__ Wg, const bf16* __restrict__ Wu, const bf16* __restrict__ Wd,
    const int* __restrict__ counts, const int* __restrict__ idxl, const float* __restrict__ wgtl,
    float* __restrict__ out)
{
    __shared__ char  tAraw[32 * TA_ROW_STRIDE];  // 65.5 KB TDM-padded activation tile
    __shared__ bf16  act[32][128 + 8];           // 8.5 KB, padded
    __shared__ int   rtok[32];
    __shared__ float rwgt[32];

    const int e   = blockIdx.y;
    const int cnt = counts[e];
    const int r0  = blockIdx.x * 32;
    if (r0 >= cnt) return;

    const int tid = threadIdx.x, w = tid >> 5, l = tid & 31;
    const int khalf = l >> 4, arow = l & 15, bcol = l & 15;

    if (tid < 32) {
        const int rr = r0 + tid;
        const bool v = rr < cnt;
        rtok[tid] = v ? idxl[e * T_TOK + rr] : 0;
        rwgt[tid] = v ? wgtl[e * T_TOK + rr] : 0.f;
    }

    // ---- stage 32 routed rows of t into LDS with two concurrent DMA paths:
    //  rows  0..15: Tensor Data Mover gather (one D#, 16-bit row indices, TENSORcnt)
    //  rows 16..31: wave-issued async global->LDS b128 copies     (ASYNCcnt)
    if (w == 0) {
        // clamped, non-decreasing row indices (gather-mode OOB requirement)
        unsigned ri[16];
        #pragma unroll
        for (int j = 0; j < 16; ++j) {
            int rr = r0 + j; if (rr >= cnt) rr = cnt - 1;
            ri[j] = (unsigned)idxl[e * T_TOK + rr] & 0xFFFFu;
        }
        const unsigned long long ga = (unsigned long long)(size_t)t;
        const unsigned ldsa = (unsigned)(size_t)&tAraw[0];   // low 32b = LDS offset
        u32x4 g0;
        g0[0] = 0x80000001u;                        // gather_mode=1, 16b idx, count=1
        g0[1] = ldsa;                               // lds_addr
        g0[2] = (unsigned)ga;                       // global_addr[31:0]
        g0[3] = (unsigned)(ga >> 32) | (2u << 30);  // global_addr[56:32] | type=2
        i32x8 g1;
        g1[0] = (1 << 16) | (1 << 20) | (7 << 22) | (5 << 25);
                 // data_size=2B | pad_enable | pad_interval=1024B | pad_amount=24B
        g1[1] = (int)((H_DIM & 0xFFFF) << 16);      // tensor_dim0[15:0] at [31:16]
        g1[2] = (int)((H_DIM >> 16) | ((T_TOK & 0xFFFF) << 16)); // dim0 hi | dim1 lo
        g1[3] = (int)((T_TOK >> 16) | (H_DIM << 16));            // dim1 hi | tile_dim0
        g1[4] = 16;                                 // tile_dim1 = #valid indices
        g1[5] = H_DIM;                              // tensor_dim0_stride[31:0]
        g1[6] = 0;
        g1[7] = 0;
        i32x4 g2, g3;
        #pragma unroll
        for (int j = 0; j < 4; ++j) {
            g2[j] = (int)(ri[2 * j]     | (ri[2 * j + 1] << 16));
            g3[j] = (int)(ri[8 + 2 * j] | (ri[8 + 2 * j + 1] << 16));
        }
        i32x8 gz = {0,0,0,0,0,0,0,0};               // unused trailing SGPR group (6-arg form)
        __builtin_amdgcn_tensor_load_to_lds(g0, g1, g2, g3, gz, 0);
        __builtin_amdgcn_s_wait_tensorcnt(0);
    } else if (w >= 4) {
        // rows 16..31, same padded layout as the TDM tile
        const int row = 16 + ((tid - 128) >> 3), seg = tid & 7;
        const int rr = r0 + row;
        const int tok = (rr < cnt) ? idxl[e * T_TOK + rr] : idxl[e * T_TOK + cnt - 1];
        const bf16* src = t + (size_t)tok * H_DIM + seg * 128;
        const unsigned dst = (unsigned)(size_t)&tAraw[0] + tA_off(row, seg * 128);
        #pragma unroll
        for (int i = 0; i < 16; ++i)
            async_copy_b128(dst + i * 16, (const char*)src + i * 16);
        wait_asynccnt0();
    }
    __syncthreads();

    const bf16* wg = Wg + (size_t)e * I_DIM * H_DIM;
    const bf16* wu = Wu + (size_t)e * I_DIM * H_DIM;
    const bf16* wd = Wd + (size_t)e * H_DIM * I_DIM;

    f32x8 dacc[2][8];
    #pragma unroll
    for (int ms = 0; ms < 2; ++ms)
        #pragma unroll
        for (int n = 0; n < 8; ++n) dacc[ms][n] = zero8();

    for (int i0 = 0; i0 < I_DIM; i0 += 128) {
        // ---- gate & up for this wave's 16 inter-cols, K = H
        const int ncol = i0 + w * 16;
        f32x8 gacc[2] = { zero8(), zero8() };
        f32x8 uacc[2] = { zero8(), zero8() };
        for (int k = 0; k < H_DIM; k += 32) {
            AFrag a0, a1;
            a0.q[0] = *(const u32x4*)(tAraw + tA_off(arow,      k     + khalf * 16));
            a0.q[1] = *(const u32x4*)(tAraw + tA_off(arow,      k + 8 + khalf * 16));
            a1.q[0] = *(const u32x4*)(tAraw + tA_off(16 + arow, k     + khalf * 16));
            a1.q[1] = *(const u32x4*)(tAraw + tA_off(16 + arow, k + 8 + khalf * 16));
            const bf16* bgp = wg + (size_t)(ncol + bcol) * H_DIM + k + khalf * 16;
            const bf16* bup = wu + (size_t)(ncol + bcol) * H_DIM + k + khalf * 16;
            bf16x16 bg = *(const bf16x16*)bgp;
            bf16x16 bu = *(const bf16x16*)bup;
            prefetch_wgp(bgp + 32);
            prefetch_wgp(bup + 32);
            gacc[0] = wmma_bf16(a0.v, bg, gacc[0]);
            gacc[1] = wmma_bf16(a1.v, bg, gacc[1]);
            uacc[0] = wmma_bf16(a0.v, bu, uacc[0]);
            uacc[1] = wmma_bf16(a1.v, bu, uacc[1]);
        }
        // ---- act = silu(g) * u  (g,u fragments have identical C layout)
        #pragma unroll
        for (int ms = 0; ms < 2; ++ms)
            #pragma unroll
            for (int r = 0; r < 8; ++r) {
                const float gv = gacc[ms][r], uv = uacc[ms][r];
                const float sv = gv / (1.f + __expf(-gv));
                act[ms * 16 + khalf * 8 + r][w * 16 + bcol] = (bf16)(sv * uv);
            }
        __syncthreads();
        // ---- down-proj: wave w -> output cols [128w, 128w+128), K = 128 chunk
        for (int kk = 0; kk < 128; kk += 32) {
            AFrag a0, a1;
            a0.q[0] = *(const u32x4*)&act[arow     ][kk     + khalf * 16];
            a0.q[1] = *(const u32x4*)&act[arow     ][kk + 8 + khalf * 16];
            a1.q[0] = *(const u32x4*)&act[16 + arow][kk     + khalf * 16];
            a1.q[1] = *(const u32x4*)&act[16 + arow][kk + 8 + khalf * 16];
            #pragma unroll
            for (int n = 0; n < 8; ++n) {
                const bf16* bdp = wd + (size_t)(w * 128 + n * 16 + bcol) * I_DIM + i0 + kk + khalf * 16;
                bf16x16 bd = *(const bf16x16*)bdp;
                prefetch_wgp(bdp + 32);
                dacc[0][n] = wmma_bf16(a0.v, bd, dacc[0][n]);
                dacc[1][n] = wmma_bf16(a1.v, bd, dacc[1][n]);
            }
        }
        __syncthreads();
    }

    // ---- scale by routing weight, accumulate into out (h already there)
    #pragma unroll
    for (int ms = 0; ms < 2; ++ms)
        #pragma unroll
        for (int n = 0; n < 8; ++n)
            #pragma unroll
            for (int r = 0; r < 8; ++r) {
                const int row = ms * 16 + khalf * 8 + r;
                const float wv = rwgt[row];
                if (wv != 0.f) {
                    const int col = w * 128 + n * 16 + bcol;
                    atomicAdd(out + (size_t)rtok[row] * H_DIM + col, dacc[ms][n][r] * wv);
                }
            }
}

// ---------------------------------------------------------------- launch
extern "C" void kernel_launch(void* const* d_in, const int* in_sizes, int n_in,
                              void* d_out, int out_size, void* d_ws, size_t ws_size,
                              hipStream_t stream) {
    const float* x   = (const float*)d_in[0];
    const float* g1  = (const float*)d_in[1];
    const float* b1  = (const float*)d_in[2];
    const float* Wa  = (const float*)d_in[3];
    const float* g2  = (const float*)d_in[4];
    const float* b2  = (const float*)d_in[5];
    const float* gw  = (const float*)d_in[6];
    const float* Wgp = (const float*)d_in[7];
    const float* Wup = (const float*)d_in[8];
    const float* Wdp = (const float*)d_in[9];
    float* out = (float*)d_out;

    char* p = (char*)d_ws;
    bf16* aW = (bf16*)p;  p += (size_t)H_DIM * H_DIM * sizeof(bf16);
    bf16* gW = (bf16*)p;  p += (size_t)NE * I_DIM * H_DIM * sizeof(bf16);
    bf16* uW = (bf16*)p;  p += (size_t)NE * I_DIM * H_DIM * sizeof(bf16);
    bf16* dW = (bf16*)p;  p += (size_t)NE * H_DIM * I_DIM * sizeof(bf16);
    bf16* tB = (bf16*)p;  p += (size_t)T_TOK * H_DIM * sizeof(bf16);
    int*   counts = (int*)p;   p += 256;
    int*   idxl   = (int*)p;   p += (size_t)NE * T_TOK * sizeof(int);
    float* wgtl   = (float*)p;

    k_init<<<1, 32, 0, stream>>>(counts);
    k_cvt<<<1024, 256, 0, stream>>>(Wa,  aW, H_DIM * H_DIM);
    k_cvt<<<4096, 256, 0, stream>>>(Wgp, gW, NE * I_DIM * H_DIM);
    k_cvt<<<4096, 256, 0, stream>>>(Wup, uW, NE * I_DIM * H_DIM);
    k_cvt<<<4096, 256, 0, stream>>>(Wdp, dW, NE * H_DIM * I_DIM);
    k_ln_attn<<<T_TOK / 32, 256, 0, stream>>>(x, g1, b1, aW, out);
    k_ln2<<<T_TOK / 8, 256, 0, stream>>>(out, g2, b2, tB);
    k_gate<<<T_TOK / 8, 256, 0, stream>>>(tB, gw, counts, idxl, wgtl);
    k_moe<<<dim3(T_TOK / 32, NE), 256, 0, stream>>>(tB, gW, uW, dW, counts, idxl, wgtl, out);
}
